// MultiHead_60971355734194
// MI455X (gfx1250) — matrix operand order repaired and checked by reading, not compiled, not run on hardware
//
#include <hip/hip_runtime.h>
#include <hip/hip_bf16.h>

#define N_NODES 50000
#define DEG     16
#define NF      32
#define H_HEADS 4
#define HF      128     // H * NF
#define SLOPE   0.2f

typedef __attribute__((ext_vector_type(16))) _Float16 v16h;
typedef __attribute__((ext_vector_type(8)))  float    v8f;

// ---- WMMA operand builders (contiguous k order, A and B matched) -------------------

// A: 16(M) x 32(K) f16, row-major in LDS with `stride` halves per row.
// lanes 0-15: M=lane, halves 0-15 = K0..15
// lanes16-31: M=lane-16, halves 0-15 = K16..31
// Contiguous 8-half groups -> two ds_load_b128 per lane.
static __device__ __forceinline__ v16h make_A(const _Float16* base, int stride, int lane) {
  const int m  = lane & 15;
  const int kb = (lane < 16) ? 0 : 16;
  v16h a;
#pragma unroll
  for (int j = 0; j < 8; ++j) {
    a[j]     = base[m * stride + kb + j];
    a[j + 8] = base[m * stride + kb + 8 + j];
  }
  return a;
}

// B: 32(K) x 16(N) f16 tile taken from a COLUMN-MAJOR LDS weight image
// cm[col * stride + k].  lanes 0-15: N=lane, halves j = K k0+j;
// lanes 16-31: halves j = K k0+16+j.  16 contiguous halves -> 2x ds_load_b128.
static __device__ __forceinline__ v16h make_Bt(const _Float16* cm, int stride,
                                               int col0, int k0, int lane) {
  const int n  = lane & 15;
  const int kb = k0 + ((lane < 16) ? 0 : 16);
  const _Float16* p = cm + (col0 + n) * stride + kb;
  v16h b;
#pragma unroll
  for (int j = 0; j < 16; ++j) b[j] = p[j];
  return b;
}

static __device__ __forceinline__ v8f wmma_f16(v16h a, v16h b, v8f c) {
  return __builtin_amdgcn_wmma_f32_16x16x32_f16(false, a, false, b, (short)0, c, false, false);
}

static __device__ __forceinline__ void lds_fence() {
  asm volatile("s_wait_dscnt 0" ::: "memory");   // CDNA5 split DS counter
  __builtin_amdgcn_wave_barrier();
}

// ---- Kernel 1: wv = h @ W_node  + per-node attention scalars au/av ----------
// block = 128 threads (4 waves); each wave computes a 16x128 wv tile.
__global__ __launch_bounds__(128) void wv_kernel(const float* __restrict__ h,
                                                 const float* __restrict__ Wn,
                                                 const float* __restrict__ watt,
                                                 float* __restrict__ wv,
                                                 float* __restrict__ au,
                                                 float* __restrict__ av) {
  __shared__ __align__(16) _Float16 sWnT[HF * NF];    // W_node col-major f16  8 KB
  __shared__ __align__(16) float    sWatt[H_HEADS * 3 * NF];
  __shared__ __align__(16) _Float16 sH[4][16 * NF];   // activation tiles      4 KB
  __shared__ __align__(16) float    sOut[4][16 * HF]; // C tiles (bounce)     32 KB

  for (int i = threadIdx.x; i < NF * HF; i += 128) {  // transpose-stage W_node
    const int k = i >> 7, c = i & 127;
    sWnT[c * NF + k] = (_Float16)Wn[i];
  }
  for (int i = threadIdx.x; i < H_HEADS * 3 * NF; i += 128) sWatt[i] = watt[i];
  __syncthreads();

  const int wave = threadIdx.x >> 5;
  const int lane = threadIdx.x & 31;
  const int n0   = blockIdx.x * 64 + wave * 16;
  const int rb   = (lane >= 16) ? 8 : 0;
  const int cl   = lane & 15;

  // stage 16 rows of h as f16 (float4 loads, b64 LDS stores)
  {
    const float4* hp = (const float4*)(h + (size_t)((n0 < N_NODES - 16) ? n0 : (N_NODES - 16)) * NF);
#pragma unroll
    for (int it = 0; it < 4; ++it) {
      const int idx = it * 32 + lane;                 // 0..127 float4s
      const float4 q = hp[idx];
      _Float16* d = &sH[wave][(idx >> 3) * NF + (idx & 7) * 4];
      d[0] = (_Float16)q.x; d[1] = (_Float16)q.y;
      d[2] = (_Float16)q.z; d[3] = (_Float16)q.w;
    }
  }
  lds_fence();

  const v16h A = make_A(sH[wave], NF, lane);
#pragma unroll
  for (int nt = 0; nt < 8; ++nt) {
    v16h B  = make_Bt(sWnT, NF, nt * 16, 0, lane);
    v8f acc = {};
    acc = wmma_f16(A, B, acc);
#pragma unroll
    for (int r = 0; r < 8; ++r)
      sOut[wave][(r + rb) * HF + nt * 16 + cl] = acc[r];
  }
  lds_fence();

  const int nbase = (n0 < N_NODES - 16) ? n0 : (N_NODES - 16);
  // coalesced wv stores: one float4 per lane per row (wave-uniform guard)
#pragma unroll
  for (int r = 0; r < 16; ++r) {
    const float4 q = *(const float4*)&sOut[wave][r * HF + lane * 4];
    *(float4*)&wv[(size_t)(nbase + r) * HF + lane * 4] = q;
  }
  // au/av: lane -> (row = lane&15, heads (lane>>4) and (lane>>4)+2)
  const int row = lane & 15;
#pragma unroll
  for (int t = 0; t < 2; ++t) {
    const int hh = (lane >> 4) + 2 * t;
    const float* wrow = &sOut[wave][row * HF + hh * NF];
    const float* wu   = &sWatt[hh * 3 * NF];
    const float* wvv  = &sWatt[hh * 3 * NF + 2 * NF];
    float su = 0.f, sv = 0.f;
#pragma unroll
    for (int f = 0; f < NF; ++f) { su += wrow[f] * wu[f]; sv += wrow[f] * wvv[f]; }
    au[(nbase + row) * H_HEADS + hh] = su;
    av[(nbase + row) * H_HEADS + hh] = sv;
  }
}

// ---- Kernel 2: fused we-GEMM + softmax + aggregate + output GEMM ------------
// 1 block (4 waves) = 16 dst nodes; each wave owns 4 nodes (16 edges each).
__global__ __launch_bounds__(128) void edge_kernel(const float* __restrict__ ef,
                                                   const int*   __restrict__ src,
                                                   const float* __restrict__ We,
                                                   const float* __restrict__ watt,
                                                   const float* __restrict__ Ws,
                                                   const float* __restrict__ bias,
                                                   const float* __restrict__ wv,
                                                   const float* __restrict__ au,
                                                   const float* __restrict__ av,
                                                   float*       __restrict__ out) {
  __shared__ __align__(16) _Float16 sWeT[HF * NF];        // W_edge col-major   8 KB
  __shared__ __align__(16) _Float16 sWsT[NF * HF];        // W_scale col-major  8 KB
  __shared__ __align__(16) float    sWatt[H_HEADS * 3 * NF];
  __shared__            float       sBias[NF];
  __shared__ __align__(16) _Float16 sE[4][16 * NF];       // edge tiles f16     4 KB
  __shared__ __align__(16) float    sT[4][16 * HF];       // we tiles f32      32 KB
  __shared__            float       sScore[4][16 * H_HEADS];
  __shared__            int         sSrc[4][16];
  __shared__ __align__(16) _Float16 sAgg[16 * HF];        // out-GEMM A tile    4 KB
  __shared__ __align__(16) float    sO[16 * NF];          // out bounce         2 KB

  for (int i = threadIdx.x; i < NF * HF; i += 128) {      // transpose W_edge
    const int k = i >> 7, c = i & 127;
    sWeT[c * NF + k] = (_Float16)We[i];
  }
  for (int i = threadIdx.x; i < HF * NF; i += 128) {      // transpose W_scale
    const int k = i >> 5, c = i & 31;
    sWsT[c * HF + k] = (_Float16)Ws[i];
  }
  for (int i = threadIdx.x; i < H_HEADS * 3 * NF; i += 128) sWatt[i] = watt[i];
  if (threadIdx.x < NF) sBias[threadIdx.x] = bias[threadIdx.x];
  __syncthreads();

  const int wave     = threadIdx.x >> 5;
  const int lane     = threadIdx.x & 31;
  const int nodeBase = blockIdx.x * 16;
  const int rb       = (lane >= 16) ? 8 : 0;
  const int cl       = lane & 15;

  // loop-invariant B operands for the we-GEMM (8 tiles, kept in VGPRs)
  v16h Bwe[8];
#pragma unroll
  for (int nt = 0; nt < 8; ++nt) Bwe[nt] = make_Bt(sWeT, NF, nt * 16, 0, lane);

  for (int i = 0; i < 4; ++i) {
    const int n  = nodeBase + wave * 4 + i;
    const int e0 = n * DEG;
    if (i < 3) __builtin_prefetch(&ef[(size_t)(e0 + DEG) * NF], 0, 3);

    // stage this node's 16x32 edge block as f16 (float4 loads, b64 LDS stores)
    {
      const float4* ep = (const float4*)(ef + (size_t)e0 * NF);
#pragma unroll
      for (int it = 0; it < 4; ++it) {
        const int idx = it * 32 + lane;
        const float4 q = ep[idx];
        _Float16* d = &sE[wave][(idx >> 3) * NF + (idx & 7) * 4];
        d[0] = (_Float16)q.x; d[1] = (_Float16)q.y;
        d[2] = (_Float16)q.z; d[3] = (_Float16)q.w;
      }
    }
    if (lane < 16) sSrc[wave][lane] = src[e0 + lane];
    lds_fence();

    // we[16,128] = eTile @ W_edge  (8 x wmma 16x16x32, B hoisted)
    const v16h A = make_A(sE[wave], NF, lane);
#pragma unroll
    for (int nt = 0; nt < 8; ++nt) {
      v8f acc = {};
      acc = wmma_f16(A, Bwe[nt], acc);
#pragma unroll
      for (int r = 0; r < 8; ++r)
        sT[wave][(r + rb) * HF + nt * 16 + cl] = acc[r];
    }
    lds_fence();

    // attention + segment softmax: the 16-edge segment == a 16-lane shfl group
    const int ee = lane & 15;
#pragma unroll
    for (int t = 0; t < 2; ++t) {
      const int hh = (lane >> 4) + 2 * t;
      const float* wae = &sWatt[hh * 3 * NF + NF];          // wa_e
      float s = au[sSrc[wave][ee] * H_HEADS + hh] + av[n * H_HEADS + hh];
#pragma unroll
      for (int f = 0; f < NF; ++f)
        s += sT[wave][ee * HF + hh * NF + f] * wae[f];
      s = (s > 0.f) ? s : s * SLOPE;                        // leaky relu
      float m = s;
#pragma unroll
      for (int off = 8; off >= 1; off >>= 1)
        m = fmaxf(m, __shfl_xor(m, off, 16));
      const float ex = __expf(s - m);
      float sum = ex;
#pragma unroll
      for (int off = 8; off >= 1; off >>= 1)
        sum += __shfl_xor(sum, off, 16);
      sScore[wave][ee * H_HEADS + hh] = ex / sum;
    }
    lds_fence();

    // agg[f] = sum_e score[e,h(f)] * wv[src[e],f] * we[e,f]   (wv hits L2)
    const int nloc = wave * 4 + i;
#pragma unroll
    for (int it = 0; it < 4; ++it) {
      const int f = it * 32 + lane;
      float acc = 0.f;
#pragma unroll
      for (int e2 = 0; e2 < 16; ++e2)
        acc += sScore[wave][e2 * H_HEADS + it] * sT[wave][e2 * HF + f]
             * wv[(size_t)sSrc[wave][e2] * HF + f];
      sAgg[nloc * HF + f] = (_Float16)acc;
    }
  }
  __syncthreads();

  // out[16,32] = agg[16,128] @ W_scale[128,32] + bias; waves 0/1 one N-tile each
  if (wave < 2) {
    v8f acc = {};
#pragma unroll
    for (int kk = 0; kk < 4; ++kk) {                        // K = 4 x 32
      v16h A = make_A(sAgg + kk * NF, HF, lane);
      v16h B = make_Bt(sWsT, HF, wave * 16, kk * NF, lane);
      acc = wmma_f16(A, B, acc);
    }
    const int c = wave * 16 + cl;
#pragma unroll
    for (int r = 0; r < 8; ++r)
      sO[(r + rb) * NF + c] = acc[r] + sBias[c];
  }
  __syncthreads();

  // coalesced output: one global_store_b128 per thread (16 rows x 8 float4)
  {
    const int row = threadIdx.x >> 3, q = threadIdx.x & 7;
    const float4 vq = *(const float4*)&sO[row * NF + q * 4];
    *(float4*)&out[(size_t)(nodeBase + row) * NF + q * 4] = vq;
  }
}

// ---- Host launcher ----------------------------------------------------------
extern "C" void kernel_launch(void* const* d_in, const int* in_sizes, int n_in,
                              void* d_out, int out_size, void* d_ws, size_t ws_size,
                              hipStream_t stream) {
  (void)in_sizes; (void)n_in; (void)out_size; (void)ws_size;
  const float* h    = (const float*)d_in[0];
  const float* e    = (const float*)d_in[1];
  const int*   src  = (const int*)  d_in[2];
  // d_in[3] (dst) unused: dst == repeat(arange(N), DEG) by construction
  const float* Wn   = (const float*)d_in[4];
  const float* We   = (const float*)d_in[5];
  const float* watt = (const float*)d_in[6];
  const float* Ws   = (const float*)d_in[7];
  const float* bias = (const float*)d_in[8];
  float* out = (float*)d_out;

  // workspace: wv [N,128] f32 (25.6 MB, L2-resident) + au/av [N,4] f32
  float* wv = (float*)d_ws;
  float* au = wv + (size_t)N_NODES * HF;
  float* av = au + (size_t)N_NODES * H_HEADS;

  wv_kernel<<<(N_NODES + 63) / 64, 128, 0, stream>>>(h, Wn, watt, wv, au, av);
  edge_kernel<<<N_NODES / 16, 128, 0, stream>>>(e, src, We, watt, Ws, bias, wv, au, av, out);
}
